// Model_2_73486890435292
// MI455X (gfx1250) — matrix operand order repaired and checked by reading, not compiled, not run on hardware
//
#include <hip/hip_runtime.h>

// ---------------------------------------------------------------------------
// Types for CDNA5 WMMA
// ---------------------------------------------------------------------------
typedef __attribute__((ext_vector_type(16))) __bf16          v16bf;
typedef __attribute__((ext_vector_type(2)))  __bf16          v2bf;
typedef __attribute__((ext_vector_type(16))) unsigned short  v16u;
typedef __attribute__((ext_vector_type(8)))  float           v8f;

// Native hardware converts (lower to v_cvt_pk_bf16_f32 on gfx1250, RTNE).
__device__ __forceinline__ unsigned short f32_to_bf16(float f) {
    return __builtin_bit_cast(unsigned short, (__bf16)f);
}
__device__ __forceinline__ unsigned int f32x2_to_bf16x2(float lo, float hi) {
    v2bf p; p[0] = (__bf16)lo; p[1] = (__bf16)hi;
    return __builtin_bit_cast(unsigned int, p);
}

// ---------------------------------------------------------------------------
// Generic tiled bf16 WMMA GEMM:  C = op(A @ B + bias)
//   A: (M,K) row-major f32 (lda, batch strideA)
//   B: (K,N) row-major f32 (ldb, batch strideB)
//   C: (M,N) row-major f32 (ldc, batch strideC)
//   flags: 1 = ReLU, 2 = accumulate into C
//
// Block: 256 threads (8 waves, wave32). Block tile BM=32 x BN=128, K-step 64.
// Each wave: one M tile x two N tiles -> 2 accumulators, 4 WMMAs per stage
// (A fragment reused across the two N tiles).
// Global->LDS staging is register double-buffered: next tile's
// global_load_b128s issue before the current tile's WMMAs.
// LDS tiles are stored in WMMA *fragment order* so each lane reads its
// 32-byte fragment with ds_load_b128:
//   A(16x32 bf16): lane = ((k>>3)&1)*16 + (m&15), e = (k&7) | ((k&16)>>1)
//   B(32x16 bf16): lane = ((k>>3)&1)*16 + (n&15), e = (k&7) | ((k&16)>>1)
// B staging works on 2k x 4n blocks: the (k,k+1) pair at one n packs into a
// single v_cvt_pk_bf16_f32 + ds_store_b32 (consecutive e, same lane row).
// Requirements: N % 128 == 0, K % 64 == 0 (true for every GEMM here);
// M may be ragged (clamped loads, guarded stores).
// ---------------------------------------------------------------------------
#define BM 32
#define BN 128
#define KS 64
#define FLAG_RELU 1
#define FLAG_ACC  2

__global__ __launch_bounds__(256) void gemm_bf16_wmma(
    const float* __restrict__ A, long long lda, long long strideA,
    const float* __restrict__ B, long long ldb, long long strideB,
    const float* __restrict__ bias,
    float* __restrict__ C, long long ldc, long long strideC,
    int M, int N, int K, int flags)
{
    const int bz = blockIdx.z;
    A += (long long)bz * strideA;
    B += (long long)bz * strideB;
    C += (long long)bz * strideC;

    const int tid  = threadIdx.x;
    const int lane = tid & 31;
    const int wave = tid >> 5;

    const int mBlock = blockIdx.y * BM;
    const int nBlock = blockIdx.x * BN;

    const int mt  = wave & 1;            // wave M tile (0..1)
    const int nt0 = (wave >> 1) * 2;     // wave N tiles nt0, nt0+1 (0..7)

    // fragment-ordered LDS: [ksub][tile][lane][e]
    __shared__ __align__(32) unsigned short Af[2][2][32][16];
    __shared__ __align__(32) unsigned short Bf[2][8][32][16];

    v8f acc0 = {}, acc1 = {};
    const int hlf = lane >> 4;
    const int l16 = lane & 15;

    // A staging: 32m x 64k = 2048 elems, 2 float4/thread: m = c>>4, k4=(c&15)*4
    // B staging: 64k x 128n = 8192 elems, 4 (2k x 4n) blocks/thread:
    //            block c: kp = c>>5 (k = 2*kp), n4 = (c&31)*4
    const int amA  = (tid >> 4);
    const int akA  = (tid & 15) * 4;
    const int amA2 = ((tid + 256) >> 4);
    const int akA2 = (tid & 15) * 4;

    float4 aR[2], bR[8];

    auto loadA = [&](int k0) {
        int gm0 = mBlock + amA;  if (gm0 >= M) gm0 = M - 1;
        int gm1 = mBlock + amA2; if (gm1 >= M) gm1 = M - 1;
        aR[0] = *(const float4*)&A[(long long)gm0 * lda + (k0 + akA)];
        aR[1] = *(const float4*)&A[(long long)gm1 * lda + (k0 + akA2)];
    };
    auto loadB = [&](int k0) {
        #pragma unroll
        for (int ci = 0; ci < 4; ++ci) {
            const int c  = tid + ci * 256;
            const int k  = (c >> 5) * 2;
            const int n4 = (c & 31) * 4;
            bR[2 * ci]     = *(const float4*)&B[(long long)(k0 + k) * ldb + (nBlock + n4)];
            bR[2 * ci + 1] = *(const float4*)&B[(long long)(k0 + k + 1) * ldb + (nBlock + n4)];
        }
    };
    auto storeTiles = [&]() {
        #pragma unroll
        for (int ci = 0; ci < 2; ++ci) {
            const int c  = tid + ci * 256;
            const int m  = c >> 4;
            const int k4 = (c & 15) * 4;
            const int ks = k4 >> 5;
            const int kk = k4 & 31;
            const int la = ((kk >> 3) & 1) * 16 + (m & 15);
            const int e0 = (kk & 7) | ((kk & 16) >> 1);     // 4 consecutive e
            ushort4 u4;
            u4.x = f32_to_bf16(aR[ci].x); u4.y = f32_to_bf16(aR[ci].y);
            u4.z = f32_to_bf16(aR[ci].z); u4.w = f32_to_bf16(aR[ci].w);
            *(ushort4*)&Af[ks][m >> 4][la][e0] = u4;        // ds_store_b64
        }
        #pragma unroll
        for (int ci = 0; ci < 4; ++ci) {
            const int c  = tid + ci * 256;
            const int k  = (c >> 5) * 2;
            const int n4 = (c & 31) * 4;
            const int ks = k >> 5;
            const int kk = k & 31;
            const int lb = ((kk >> 3) & 1) * 16;
            const int e  = (kk & 7) | ((kk & 16) >> 1);     // even
            const float4 lo = bR[2 * ci], hi = bR[2 * ci + 1];
            *(unsigned int*)&Bf[ks][(n4 + 0) >> 4][lb + ((n4 + 0) & 15)][e] = f32x2_to_bf16x2(lo.x, hi.x);
            *(unsigned int*)&Bf[ks][(n4 + 1) >> 4][lb + ((n4 + 1) & 15)][e] = f32x2_to_bf16x2(lo.y, hi.y);
            *(unsigned int*)&Bf[ks][(n4 + 2) >> 4][lb + ((n4 + 2) & 15)][e] = f32x2_to_bf16x2(lo.z, hi.z);
            *(unsigned int*)&Bf[ks][(n4 + 3) >> 4][lb + ((n4 + 3) & 15)][e] = f32x2_to_bf16x2(lo.w, hi.w);
        }
    };

    loadA(0);
    loadB(0);

    for (int k0 = 0; k0 < K; k0 += KS) {
        storeTiles();
        __syncthreads();

        if (k0 + KS < K) {               // prefetch next tile (uniform branch)
            loadA(k0 + KS);
            loadB(k0 + KS);
        }

        #pragma unroll
        for (int ks = 0; ks < 2; ++ks) {
            v16u au  = *(const v16u*)&Af[ks][mt][lane][0];
            v16u bu0 = *(const v16u*)&Bf[ks][nt0][lane][0];
            v16u bu1 = *(const v16u*)&Bf[ks][nt0 + 1][lane][0];
            acc0 = __builtin_amdgcn_wmma_f32_16x16x32_bf16(
                       false, __builtin_bit_cast(v16bf, au),
                       false, __builtin_bit_cast(v16bf, bu0),
                       (short)0, acc0, false, false);
            acc1 = __builtin_amdgcn_wmma_f32_16x16x32_bf16(
                       false, __builtin_bit_cast(v16bf, au),
                       false, __builtin_bit_cast(v16bf, bu1),
                       (short)0, acc1, false, false);
        }
        __syncthreads();
    }

    // epilogue: bias + (acc) + relu; C/D layout: row = r + 8*half, col = lane%16
    const int col0 = nBlock + nt0 * 16 + l16;
    #pragma unroll
    for (int r = 0; r < 8; ++r) {
        int row = mBlock + mt * 16 + r + hlf * 8;
        if (row < M) {
            float v0 = acc0[r], v1 = acc1[r];
            if (bias) { v0 += bias[col0]; v1 += bias[col0 + 16]; }
            float* d0 = &C[(long long)row * ldc + col0];
            float* d1 = d0 + 16;
            if (flags & FLAG_ACC) { v0 += *d0; v1 += *d1; }
            if (flags & FLAG_RELU) { v0 = fmaxf(v0, 0.0f); v1 = fmaxf(v1, 0.0f); }
            *d0 = v0;
            *d1 = v1;
        }
    }
}

// ---------------------------------------------------------------------------
// split-K reduce: C[i] = op( sum_c P[c*elems + i] + bias[i % N] )
// ---------------------------------------------------------------------------
__global__ void splitk_reduce_kernel(const float* __restrict__ P,
                                     const float* __restrict__ bias,
                                     float* __restrict__ C,
                                     int chunks, int elems, int N, int flags) {
    int idx = blockIdx.x * blockDim.x + threadIdx.x;
    if (idx >= elems) return;
    float s = 0.0f;
    for (int c = 0; c < chunks; ++c) s += P[(long long)c * elems + idx];
    if (bias) s += bias[idx % N];
    if (flags & FLAG_RELU) s = fmaxf(s, 0.0f);
    C[idx] = s;
}

// ---------------------------------------------------------------------------
// pos_encoding: P[k, 2i] = sin(k / 10000^(2i/d)); P[k, 2i+1] = cos(...)
// ---------------------------------------------------------------------------
__global__ void pos_encoding_kernel(float* __restrict__ P, int seq, int d) {
    int idx = blockIdx.x * blockDim.x + threadIdx.x;
    if (idx >= seq * d) return;
    int k = idx / d;
    int j = idx - k * d;
    int i = j >> 1;
    float ang = (float)k * __expf(-(2.0f * (float)i / (float)d) * 9.210340371976184f);
    P[idx] = (j & 1) ? __cosf(ang) : __sinf(ang);
}

// ---------------------------------------------------------------------------
// row_sum: rs[row] = sum_j H[row, j]   (one wave32 per row)
// ---------------------------------------------------------------------------
__global__ void row_sum_kernel(const float* __restrict__ H, float* __restrict__ rs,
                               int rows, int cols) {
    int row  = blockIdx.x * (blockDim.x >> 5) + (threadIdx.x >> 5);
    int lane = threadIdx.x & 31;
    if (row >= rows) return;
    float s = 0.0f;
    for (int j = lane; j < cols; j += 32) s += H[(long long)row * cols + j];
    #pragma unroll
    for (int off = 16; off > 0; off >>= 1) s += __shfl_down(s, off, 32);
    if (lane == 0) rs[row] = s;
}

// ---------------------------------------------------------------------------
// broadcast add: H[b,s,:] += Pos[s,:]
// ---------------------------------------------------------------------------
__global__ void bcast_add_kernel(float* __restrict__ H, const float* __restrict__ Pos,
                                 int Bn, int S, int D) {
    long long idx   = (long long)blockIdx.x * blockDim.x + threadIdx.x;
    long long total = (long long)Bn * S * D;
    if (idx >= total) return;
    long long rem = idx % ((long long)S * D);
    int s = (int)(rem / D);
    int j = (int)(rem - (long long)s * D);
    H[idx] += Pos[s * D + j];
}

// ---------------------------------------------------------------------------
// Host-side launcher
// ---------------------------------------------------------------------------
static inline void launch_gemm(const float* A, long long lda, long long sA,
                               const float* B, long long ldb, long long sB,
                               const float* bias,
                               float* C, long long ldc, long long sC,
                               int M, int N, int K, int flags, int batch,
                               hipStream_t stream) {
    dim3 grid(N / BN, (M + BM - 1) / BM, batch);
    gemm_bf16_wmma<<<grid, 256, 0, stream>>>(A, lda, sA, B, ldb, sB, bias,
                                             C, ldc, sC, M, N, K, flags);
}

extern "C" void kernel_launch(void* const* d_in, const int* in_sizes, int n_in,
                              void* d_out, int out_size, void* d_ws, size_t ws_size,
                              hipStream_t stream) {
    const int Bn = 16, S = 128, V = 32000, E = 128, R1 = 256, R2 = 640, L = 512;

    const float* x       = (const float*)d_in[0];
    const float* W_emb   = (const float*)d_in[1];
    const float* b_emb   = (const float*)d_in[2];
    const float* W_pos1  = (const float*)d_in[3];
    const float* b_pos1  = (const float*)d_in[4];
    const float* W_red   = (const float*)d_in[5];
    const float* b_red   = (const float*)d_in[6];
    const float* W_pos2  = (const float*)d_in[7];
    const float* b_pos2  = (const float*)d_in[8];
    const float* W_red2  = (const float*)d_in[9];
    const float* b_red2  = (const float*)d_in[10];
    const float* W_pos3  = (const float*)d_in[11];
    const float* b_pos3  = (const float*)d_in[12];
    const float* W_down2 = (const float*)d_in[13];
    const float* b_down2 = (const float*)d_in[14];
    const float* W_flat  = (const float*)d_in[15];
    const float* b_flat  = (const float*)d_in[16];
    const float* W_d1    = (const float*)d_in[17];
    const float* b_d1    = (const float*)d_in[18];
    const float* W_d2    = (const float*)d_in[19];
    const float* b_d2    = (const float*)d_in[20];
    const float* W_d3    = (const float*)d_in[21];
    const float* b_d3    = (const float*)d_in[22];
    const float* W_p1    = (const float*)d_in[23];
    const float* b_p1    = (const float*)d_in[24];
    const float* W_p2    = (const float*)d_in[25];
    const float* b_p2    = (const float*)d_in[26];

    float* out = (float*)d_out;   // (16, 32000)

    // ---- workspace carve-up (all f32) ----
    const int KCH = 32;                                  // split-K chunks for W_flat
    float* ws = (float*)d_ws;
    float* h1  = ws; ws += (long long)Bn * S * E;        // 2048 x 128
    float* pe  = ws; ws += (long long)S * 2 * R2;        // up to 128 x 1280
    float* pos = ws; ws += (long long)S * R2;            // up to 128 x 640
    float* rs  = ws; ws += (long long)Bn * S;            // 2048
    float* M1  = ws; ws += (long long)Bn * E * R1;       // 16 x 32768
    float* h2  = ws; ws += (long long)Bn * S * R1;       // 2048 x 256
    float* M2  = ws; ws += (long long)Bn * R1 * R2;      // 16 x 163840
    float* h3  = ws; ws += (long long)Bn * S * R2;       // 2048 x 640
    float* h4  = ws; ws += (long long)Bn * S * L;        // 2048 x 512
    float* pk  = ws; ws += (long long)KCH * Bn * L;      // 32 x (16 x 512) partials
    float* h5  = ws; ws += (long long)Bn * L;            // 16 x 512
    float* h6  = ws; ws += (long long)Bn * L;
    float* h7  = ws; ws += (long long)Bn * L;
    float* h8  = ws; ws += (long long)Bn * L;

    // ---- 1) h1 = relu(x @ W_emb + b_emb)   [dominant: x read once, 262 MB] ----
    launch_gemm(x, V, 0, W_emb, E, 0, b_emb, h1, E, 0,
                Bn * S, E, V, FLAG_RELU, 1, stream);

    // ---- 2) pos1 = relu(pe(S,2E) @ W_pos1 + b_pos1); h1 += pos1 ----
    pos_encoding_kernel<<<(S * 2 * E + 255) / 256, 256, 0, stream>>>(pe, S, 2 * E);
    launch_gemm(pe, 2 * E, 0, W_pos1, E, 0, b_pos1, pos, E, 0,
                S, E, 2 * E, FLAG_RELU, 1, stream);
    bcast_add_kernel<<<((long long)Bn * S * E + 255) / 256, 256, 0, stream>>>(h1, pos, Bn, S, E);

    // ---- 3) square-memory 1 (factorized) ----
    row_sum_kernel<<<(Bn * S + 7) / 8, 256, 0, stream>>>(h1, rs, Bn * S, E);
    launch_gemm(rs, S, 0, W_red, (long long)E * R1, 0, nullptr,
                M1, (long long)E * R1, 0, Bn, E * R1, S, 0, 1, stream);
    launch_gemm(h1, E, (long long)S * E, M1, R1, (long long)E * R1, b_red,
                h2, R1, (long long)S * R1, S, R1, E, FLAG_RELU, Bn, stream);

    // ---- 4) pos2 ----
    pos_encoding_kernel<<<(S * 2 * R1 + 255) / 256, 256, 0, stream>>>(pe, S, 2 * R1);
    launch_gemm(pe, 2 * R1, 0, W_pos2, R1, 0, b_pos2, pos, R1, 0,
                S, R1, 2 * R1, FLAG_RELU, 1, stream);
    bcast_add_kernel<<<((long long)Bn * S * R1 + 255) / 256, 256, 0, stream>>>(h2, pos, Bn, S, R1);

    // ---- 5) square-memory 2 (factorized) ----
    row_sum_kernel<<<(Bn * S + 7) / 8, 256, 0, stream>>>(h2, rs, Bn * S, R1);
    launch_gemm(rs, S, 0, W_red2, (long long)R1 * R2, 0, nullptr,
                M2, (long long)R1 * R2, 0, Bn, R1 * R2, S, 0, 1, stream);
    launch_gemm(h2, R1, (long long)S * R1, M2, R2, (long long)R1 * R2, b_red2,
                h3, R2, (long long)S * R2, S, R2, R1, FLAG_RELU, Bn, stream);

    // ---- 6) pos3 ----
    pos_encoding_kernel<<<(S * 2 * R2 + 255) / 256, 256, 0, stream>>>(pe, S, 2 * R2);
    launch_gemm(pe, 2 * R2, 0, W_pos3, R2, 0, b_pos3, pos, R2, 0,
                S, R2, 2 * R2, FLAG_RELU, 1, stream);
    bcast_add_kernel<<<((long long)Bn * S * R2 + 255) / 256, 256, 0, stream>>>(h3, pos, Bn, S, R2);

    // ---- 7) h4 = relu(h3 @ W_down2 + b_down2)  (2048,640)@(640,512) ----
    launch_gemm(h3, R2, 0, W_down2, L, 0, b_down2, h4, L, 0,
                Bn * S, L, R2, FLAG_RELU, 1, stream);

    // ---- 8) h5 = relu(h4.reshape(16,65536) @ W_flat + b_flat)  [134 MB W_flat]
    //         split-K over 32 chunks via the batch dimension, then reduce ----
    {
        const int KC = (S * L) / KCH;                    // 2048 per chunk
        launch_gemm(h4, (long long)S * L, KC,            // A: advance 2048 cols per chunk
                    W_flat, L, (long long)KC * L,        // B: advance 2048 rows per chunk
                    nullptr, pk, L, (long long)Bn * L,   // C: per-chunk partials
                    Bn, L, KC, 0, KCH, stream);
        splitk_reduce_kernel<<<(Bn * L + 255) / 256, 256, 0, stream>>>(
            pk, b_flat, h5, KCH, Bn * L, L, FLAG_RELU);
    }

    // ---- 9) dense head + accumulated projections into out ----
    launch_gemm(h5, L, 0, W_d1, L, 0, b_d1, h6, L, 0, Bn, L, L, FLAG_RELU, 1, stream);
    launch_gemm(h6, L, 0, W_p1, V, 0, b_p1, out, V, 0, Bn, V, L, 0, 1, stream);           // p1 (write)
    launch_gemm(h6, L, 0, W_d2, L, 0, b_d2, h7, L, 0, Bn, L, L, FLAG_RELU, 1, stream);
    launch_gemm(h7, L, 0, W_p2, V, 0, b_p2, out, V, 0, Bn, V, L, FLAG_ACC, 1, stream);    // += p2
    launch_gemm(h7, L, 0, W_d3, L, 0, b_d3, h8, L, 0, Bn, L, L, FLAG_RELU, 1, stream);
    launch_gemm(h8, L, 0, W_p1, V, 0, b_p1, out, V, 0, Bn, V, L, FLAG_ACC, 1, stream);    // += p3
    (void)in_sizes; (void)n_in; (void)out_size; (void)ws_size;
}
